// MaskedDeepGRU_2911987826992
// MI455X (gfx1250) — matrix operand order repaired and checked by reading, not compiled, not run on hardware
//
#include <hip/hip_runtime.h>

#define SEQ   512
#define NB    64
#define INDIM 512
#define HDIM  512
#define GDIM  1536   // 3*HDIM
#define LDST  520    // padded LDS row stride in halves (260 dwords % 64 banks = 4)

typedef __attribute__((ext_vector_type(16))) _Float16 v16h;
typedef __attribute__((ext_vector_type(8)))  _Float16 v8h;
typedef __attribute__((ext_vector_type(8)))  float    v8f;

// ---------------------------------------------------------------------------
// WMMA fragment loaders (16-bit operands, contiguous k order, wave32)
// A 16x32 f16: lane<16 holds row M=lane, K = k0..k0+15
//              lane>=16 holds row M=lane-16, K = k0+16..k0+31
// ---------------------------------------------------------------------------
static __device__ __forceinline__ v16h load_a_frag(const _Float16* __restrict__ base,
                                                   int ldk, int m0, int k0, int lane) {
  const int row = m0 + (lane & 15);
  const int kb  = k0 + ((lane >> 4) << 4);           // +16 for upper half-wave
  const _Float16* p = base + (size_t)row * ldk + kb;
  v8h lo = *(const v8h*)(p);                          // 16B aligned
  v8h hi = *(const v8h*)(p + 8);
  v16h a;
#pragma unroll
  for (int i = 0; i < 8; ++i) { a[i] = lo[i]; a[i + 8] = hi[i]; }
  return a;
}

// B 32x16 f16 from row-major W[G, K]: B(k, n) = W[n, k].
// lane<16: col N=lane, K = k0..k0+15 ; lane>=16: col N=lane-16, K = k0+16..k0+31
static __device__ __forceinline__ v16h load_b_frag(const _Float16* __restrict__ w,
                                                   int ldk, int n0, int k0, int lane) {
  const int row = n0 + (lane & 15);
  const int kb  = k0 + ((lane & 16) ? 16 : 0);
  return *(const v16h*)(w + (size_t)row * ldk + kb);  // 32B aligned
}

// B fragment from the padded LDS weight tile (rows are gate-local)
static __device__ __forceinline__ v16h load_b_lds(const _Float16* lds, int row0,
                                                  int k0, int lane) {
  const int row = row0 + (lane & 15);
  const int kb  = k0 + ((lane & 16) ? 16 : 0);
  const _Float16* p = lds + row * LDST + kb;          // 16B aligned
  v8h lo = *(const v8h*)(p);
  v8h hi = *(const v8h*)(p + 8);
  v16h b;
#pragma unroll
  for (int i = 0; i < 8; ++i) { b[i] = lo[i]; b[i + 8] = hi[i]; }
  return b;
}

static __device__ __forceinline__ v8f wmma_f16(v16h a, v16h b, v8f c) {
  return __builtin_amdgcn_wmma_f32_16x16x32_f16(false, a, false, b, (short)0, c,
                                                false, false);
}

// ---------------------------------------------------------------------------
// Prep kernels
// ---------------------------------------------------------------------------
__global__ void k_f32_to_f16(const float* __restrict__ in, _Float16* __restrict__ out,
                             int n) {
  for (int i = blockIdx.x * blockDim.x + threadIdx.x; i < n;
       i += gridDim.x * blockDim.x)
    out[i] = (_Float16)in[i];
}

__global__ void k_prep_w(const float* __restrict__ w, const unsigned char* __restrict__ m,
                         _Float16* __restrict__ out, int n) {
  for (int i = blockIdx.x * blockDim.x + threadIdx.x; i < n;
       i += gridDim.x * blockDim.x)
    out[i] = (_Float16)(m[i] ? w[i] : 0.0f);
}

__global__ void k_reset(_Float16* __restrict__ h, int n, unsigned int* __restrict__ ctr) {
  if (blockIdx.x == 0 && threadIdx.x == 0) *ctr = 0u;
  for (int i = blockIdx.x * blockDim.x + threadIdx.x; i < n;
       i += gridDim.x * blockDim.x)
    h[i] = (_Float16)0.0f;
}

__global__ void k_final(const _Float16* __restrict__ h, float* __restrict__ out, int n) {
  for (int i = blockIdx.x * blockDim.x + threadIdx.x; i < n;
       i += gridDim.x * blockDim.x)
    out[i] = (float)h[i];
}

// ---------------------------------------------------------------------------
// Batched input-gate GEMM:  IG[M, G] = A[M, 512] @ W[G, 512]^T   (f16 in/out,
// f32 accumulate).  One wave computes a 16(M) x 64(N) strip = 4 WMMA tiles,
// sharing the A fragment across the 4 N-tiles.
// ---------------------------------------------------------------------------
__global__ __launch_bounds__(256) void k_gemm_ig(const _Float16* __restrict__ A,
                                                 const _Float16* __restrict__ W,
                                                 _Float16* __restrict__ IG) {
  const int lane = threadIdx.x & 31;
  const int wid  = blockIdx.x * (blockDim.x >> 5) + (threadIdx.x >> 5);
  const int NG   = GDIM / 64;          // 24
  const int mt   = wid / NG;
  const int ng   = wid % NG;
  const int m0   = mt * 16;
  const int n0   = ng * 64;

  v8f acc[4] = {v8f{}, v8f{}, v8f{}, v8f{}};
  for (int k0 = 0; k0 < INDIM; k0 += 32) {
    v16h a = load_a_frag(A, INDIM, m0, k0, lane);
#pragma unroll
    for (int j = 0; j < 4; ++j) {
      v16h b = load_b_frag(W, INDIM, n0 + j * 16, k0, lane);
      acc[j] = wmma_f16(a, b, acc[j]);
    }
  }
  const int rbase = m0 + ((lane & 16) ? 8 : 0);
  const int col   = lane & 15;
#pragma unroll
  for (int j = 0; j < 4; ++j) {
    _Float16* out = IG + (size_t)rbase * GDIM + (n0 + j * 16 + col);
#pragma unroll
    for (int r = 0; r < 8; ++r) out[(size_t)r * GDIM] = (_Float16)acc[j][r];
  }
}

// ---------------------------------------------------------------------------
// Persistent GRU recurrence: 16 co-resident blocks run all SEQ timesteps in
// one launch.  Block owns 32 H-columns; its Wh slice (3 gates x 32 rows x 512)
// is staged into LDS ONCE (padded rows -> conflict-free ds_load_b128).
// Cross-block step barrier: monotonic device-scope atomic counter.
// Wave w: m_tile = w&3 (batch rows), h_tile = w>>2; the 3 gate tiles for its
// (m,h) tile stay wave-local so the nonlinearity needs no data exchange.
// ---------------------------------------------------------------------------
__global__ __launch_bounds__(256) void k_gru_seq(_Float16* __restrict__ Hbuf,
                                                 const _Float16* __restrict__ IG,
                                                 const _Float16* __restrict__ Wh,
                                                 const float* __restrict__ b_ih,
                                                 const float* __restrict__ b_hh,
                                                 _Float16* __restrict__ Y,
                                                 unsigned int* ctr) {
  __shared__ _Float16 ldsWh[96 * LDST];   // ~99.8 KB of the 320 KB WGP LDS

  const int lane = threadIdx.x & 31;
  const int w    = threadIdx.x >> 5;             // 0..7
  const int m0   = (w & 3) * 16;                 // batch tile (NB=64 -> 4 tiles)
  const int ht   = w >> 2;                       // 0..1 (h tile within block)
  const int c    = blockIdx.x * 32;              // block's H-column base

  // ---- stage Wh slice: LDS row (g*32 + r) <- Wh[g*HDIM + c + r, :] ----
  for (int idx = threadIdx.x; idx < 96 * 64; idx += blockDim.x) {
    const int row = idx >> 6;                    // 0..95
    const int seg = idx & 63;                    // 64 x 8-half segments
    const int g   = row >> 5;
    const int r   = row & 31;
    const _Float16* src = Wh + (size_t)(g * HDIM + c + r) * HDIM + seg * 8;
    *(v8h*)(&ldsWh[row * LDST + seg * 8]) = *(const v8h*)src;
  }
  __syncthreads();

  // ---- loop-invariant per-lane gate constants ----
  const int hc    = c + ht * 16 + (lane & 15);   // global H column
  const int rbase = m0 + ((lane & 16) ? 8 : 0);
  const float br  = b_ih[hc]            + b_hh[hc];
  const float bi  = b_ih[hc + HDIM]     + b_hh[hc + HDIM];
  const float bin = b_ih[hc + 2 * HDIM];
  const float bhn = b_hh[hc + 2 * HDIM];
  const int rloc  = ht * 16;                     // gate-local LDS row base

  for (int t = 0; t < SEQ; ++t) {
    const _Float16* Hin  = Hbuf + (size_t)(t & 1) * NB * HDIM;
    _Float16*       Hout = Hbuf + (size_t)((t + 1) & 1) * NB * HDIM;
    const _Float16* IG_t = IG + (size_t)t * NB * GDIM;
    _Float16*       Y_t  = Y + (size_t)t * NB * HDIM;

    v8f accR = v8f{}, accI = v8f{}, accN = v8f{};
    for (int k0 = 0; k0 < HDIM; k0 += 32) {
      v16h a  = load_a_frag(Hin, HDIM, m0, k0, lane);
      v16h bR = load_b_lds(ldsWh,            rloc, k0, lane);
      v16h bI = load_b_lds(ldsWh + 32 * LDST, rloc, k0, lane);
      v16h bN = load_b_lds(ldsWh + 64 * LDST, rloc, k0, lane);
      accR = wmma_f16(a, bR, accR);
      accI = wmma_f16(a, bI, accI);
      accN = wmma_f16(a, bN, accN);
    }

#pragma unroll
    for (int r = 0; r < 8; ++r) {
      const int m = rbase + r;
      const _Float16* igp = IG_t + (size_t)m * GDIM;
      const float igr = (float)igp[hc];
      const float igi = (float)igp[hc + HDIM];
      const float ign = (float)igp[hc + 2 * HDIM];

      const float reset = 1.0f / (1.0f + __expf(-(accR[r] + igr + br)));
      const float inp   = 1.0f / (1.0f + __expf(-(accI[r] + igi + bi)));
      const float nval  = tanhf(ign + bin + reset * (accN[r] + bhn));
      const float hold  = (float)Hin[(size_t)m * HDIM + hc];
      const float hnew  = nval + inp * (hold - nval);

      Hout[(size_t)m * HDIM + hc] = (_Float16)hnew;
      Y_t[(size_t)m * HDIM + hc]  = (_Float16)hnew;
    }

    // ---- device-scope step barrier (monotonic counter, no reset races) ----
    __threadfence();                 // publish this thread's h/Y stores
    __syncthreads();                 // whole block has published
    if (threadIdx.x == 0) {
      __hip_atomic_fetch_add(ctr, 1u, __ATOMIC_RELEASE, __HIP_MEMORY_SCOPE_AGENT);
      const unsigned int target = gridDim.x * (unsigned int)(t + 1);
      while (__hip_atomic_load(ctr, __ATOMIC_ACQUIRE, __HIP_MEMORY_SCOPE_AGENT) <
             target)
        __builtin_amdgcn_s_sleep(1);
    }
    __syncthreads();                 // block waits on the spinning wave
  }
}

// ---------------------------------------------------------------------------
// Host-side orchestration
// ---------------------------------------------------------------------------
extern "C" void kernel_launch(void* const* d_in, const int* in_sizes, int n_in,
                              void* d_out, int out_size, void* d_ws, size_t ws_size,
                              hipStream_t stream) {
  const float* x     = (const float*)d_in[0];
  const float* W_ih0 = (const float*)d_in[1];
  const float* W_hh0 = (const float*)d_in[2];
  const float* b_ih0 = (const float*)d_in[3];
  const float* b_hh0 = (const float*)d_in[4];
  const float* W_ih1 = (const float*)d_in[5];
  const float* W_hh1 = (const float*)d_in[6];
  const float* b_ih1 = (const float*)d_in[7];
  const float* b_hh1 = (const float*)d_in[8];
  const unsigned char* m_ih0 = (const unsigned char*)d_in[9];
  const unsigned char* m_hh0 = (const unsigned char*)d_in[10];
  const unsigned char* m_ih1 = (const unsigned char*)d_in[11];
  const unsigned char* m_hh1 = (const unsigned char*)d_in[12];

  char* ws = (char*)d_ws;
  size_t off = 0;
  _Float16* Xf16 = (_Float16*)(ws + off); off += (size_t)SEQ * NB * INDIM * 2;  // 32 MB
  _Float16* Wi0  = (_Float16*)(ws + off); off += (size_t)GDIM * INDIM * 2;
  _Float16* Wh0  = (_Float16*)(ws + off); off += (size_t)GDIM * HDIM * 2;
  _Float16* Wi1  = (_Float16*)(ws + off); off += (size_t)GDIM * HDIM * 2;
  _Float16* Wh1  = (_Float16*)(ws + off); off += (size_t)GDIM * HDIM * 2;
  _Float16* IG   = (_Float16*)(ws + off); off += (size_t)SEQ * NB * GDIM * 2;   // 96 MB
  _Float16* Hbuf = (_Float16*)(ws + off); off += (size_t)2 * NB * HDIM * 2;     // ping-pong
  unsigned int* ctr = (unsigned int*)(ws + off); off += 256;

  const int nX  = SEQ * NB * INDIM;
  const int nW  = GDIM * INDIM;
  const int nH2 = 2 * NB * HDIM;

  // --- prep: fp16 conversion + masked weights ---
  k_f32_to_f16<<<4096, 256, 0, stream>>>(x, Xf16, nX);
  k_prep_w<<<1024, 256, 0, stream>>>(W_ih0, m_ih0, Wi0, nW);
  k_prep_w<<<1024, 256, 0, stream>>>(W_hh0, m_hh0, Wh0, nW);
  k_prep_w<<<1024, 256, 0, stream>>>(W_ih1, m_ih1, Wi1, nW);
  k_prep_w<<<1024, 256, 0, stream>>>(W_hh1, m_hh1, Wh1, nW);

  const int gemm_blocks = (SEQ * NB / 16) * (GDIM / 64) / 8;  // 6144

  // ================= layer 0 =================
  k_reset<<<64, 256, 0, stream>>>(Hbuf, nH2, ctr);
  k_gemm_ig<<<gemm_blocks, 256, 0, stream>>>(Xf16, Wi0, IG);
  // layer-0 outputs overwrite Xf16 (x is fully consumed by the GEMM above)
  k_gru_seq<<<16, 256, 0, stream>>>(Hbuf, IG, Wh0, b_ih0, b_hh0, Xf16, ctr);

  // ================= layer 1 =================
  k_reset<<<64, 256, 0, stream>>>(Hbuf, nH2, ctr);
  k_gemm_ig<<<gemm_blocks, 256, 0, stream>>>(Xf16 /* = Y0 */, Wi1, IG);
  // layer-1 Y history is unused; recycle Xf16 as the sink
  k_gru_seq<<<16, 256, 0, stream>>>(Hbuf, IG, Wh1, b_ih1, b_hh1, Xf16, ctr);

  // after t = SEQ-1, the final hidden state lives at parity (SEQ & 1) == 0
  k_final<<<64, 256, 0, stream>>>(Hbuf, (float*)d_out, NB * HDIM);
}
